// XCiT_30133490549475
// MI455X (gfx1250) — matrix operand order repaired and checked by reading, not compiled, not run on hardware
//
#include <hip/hip_runtime.h>
#include <stdint.h>

// ---------------------------------------------------------------------------
// Types
// ---------------------------------------------------------------------------
typedef __bf16    v16bf __attribute__((ext_vector_type(16)));
typedef float     v8f   __attribute__((ext_vector_type(8)));
typedef uint32_t  v4u   __attribute__((ext_vector_type(4)));
typedef int       v8i   __attribute__((ext_vector_type(8)));
typedef int       v4i   __attribute__((ext_vector_type(4)));

#if defined(__gfx1250__) && __has_builtin(__builtin_amdgcn_tensor_load_to_lds)
#define USE_TDM 1
#else
#define USE_TDM 0
#endif

__device__ __forceinline__ float gelu_f(float x) {
    return 0.5f * x * (1.0f + erff(x * 0.70710678118654752f));
}
__device__ __forceinline__ uint16_t f2bf(float f) {
    uint32_t u = __float_as_uint(f);
    uint32_t r = u + 0x7FFFu + ((u >> 16) & 1u);
    return (uint16_t)(r >> 16);
}

#if USE_TDM
// Build a TDM 2D-tile descriptor (D# group0/group1) and issue tensor_load_to_lds.
// data_size = 2 bytes (bf16). tensor dims are the remaining extent from the tile
// start so TDM zero-fills M/N/K tails in LDS.
__device__ __forceinline__ void tdm_load_2d(uint32_t lds_off, const void* g,
                                            uint32_t td0, uint32_t td1,
                                            uint32_t tl0, uint32_t tl1,
                                            uint64_t stride_elems)
{
    uint64_t ga = (uint64_t)(uintptr_t)g;
    v4u g0;
    g0[0] = 1u;                                              // count=1, user desc
    g0[1] = lds_off;                                         // lds_addr
    g0[2] = (uint32_t)ga;                                    // global_addr[31:0]
    g0[3] = (uint32_t)((ga >> 32) & 0x1FFFFFFu) | (2u << 30);// ga[56:32] | type=2
    v8i g1;
    g1[0] = (int)(1u << 16);                                 // wg_mask=0, data_size=1 (2B)
    g1[1] = (int)((td0 & 0xFFFFu) << 16);                    // tensor_dim0 lo
    g1[2] = (int)((td0 >> 16) | ((td1 & 0xFFFFu) << 16));    // dim0 hi | dim1 lo
    g1[3] = (int)((td1 >> 16) | (tl0 << 16));                // dim1 hi | tile_dim0
    g1[4] = (int)(tl1 & 0xFFFFu);                            // tile_dim1, tile_dim2=0
    g1[5] = (int)(uint32_t)(stride_elems & 0xFFFFFFFFull);   // dim0_stride lo
    g1[6] = (int)(uint32_t)((stride_elems >> 32) & 0xFFFFu); // dim0_stride hi
    g1[7] = 0;
    v4i gz4; gz4[0] = 0; gz4[1] = 0; gz4[2] = 0; gz4[3] = 0;
    v8i gz8;
    for (int i = 0; i < 8; i++) gz8[i] = 0;
    __builtin_amdgcn_tensor_load_to_lds(g0, g1, gz4, gz4, gz8, 0);
}
#endif

// ---------------------------------------------------------------------------
// WMMA bf16 GEMM:  C[M,N] = A[M,K](bf16) * Bt[N,K](bf16, transposed), fp32 acc.
// Epilogue: (+bias) -> BN fold -> (+residual) -> gelu -> (*scale); fp32/bf16 out.
// Block = 256 threads (8 waves), block tile 128x128, K step 32, double-buffered
// LDS tiles fed by the Tensor Data Mover (wave0 leader, s_wait_tensorcnt).
// Wave grid 4(M) x 2(N); per wave 2x4 tiles of 16x16.
// ---------------------------------------------------------------------------
__global__ __launch_bounds__(256) void k_gemm_bf16_wmma(
    const uint16_t* __restrict__ A, const uint16_t* __restrict__ Bt,
    float* __restrict__ Cf, uint16_t* __restrict__ Cbf,
    int M, int N, int K,
    const float* __restrict__ bias,
    const float* __restrict__ bng, const float* __restrict__ bnb,
    const float* __restrict__ bnm, const float* __restrict__ bnv,
    const float* __restrict__ resid, const float* __restrict__ scale,
    int do_gelu)
{
    __shared__ __align__(16) uint16_t sA[2][128 * 32];
    __shared__ __align__(16) uint16_t sB[2][128 * 32];   // B tile rows are N, cols K

    const int tid  = (int)threadIdx.x;
    const int lane = tid & 31;
    const int wave = tid >> 5;
    const int wm   = wave & 3;       // 0..3 along M
    const int wn   = wave >> 2;      // 0..1 along N
    const int g    = lane >> 4;      // lane half
    const int r    = lane & 15;
    const int m0   = (int)blockIdx.y * 128;
    const int n0   = (int)blockIdx.x * 128;
    const int nk   = (K + 31) >> 5;

    v8f acc[2][4];
    v8f zf;
#pragma unroll
    for (int e = 0; e < 8; e++) zf[e] = 0.0f;
#pragma unroll
    for (int t = 0; t < 2; t++)
#pragma unroll
        for (int u = 0; u < 4; u++) acc[t][u] = zf;

#if USE_TDM
    const bool leader = (__builtin_amdgcn_readfirstlane(wave) == 0);
    uint32_t ldsAo[2], ldsBo[2];
    ldsAo[0] = (uint32_t)(uintptr_t)&sA[0][0];   // low 32 bits of flat addr = LDS offset
    ldsAo[1] = (uint32_t)(uintptr_t)&sA[1][0];
    ldsBo[0] = (uint32_t)(uintptr_t)&sB[0][0];
    ldsBo[1] = (uint32_t)(uintptr_t)&sB[1][0];
    auto issue = [&](int stage, int buf) {
        int kk0 = stage * 32;
        // A tile: 128 rows x 32 K
        tdm_load_2d(ldsAo[buf], A + (size_t)m0 * K + kk0,
                    (uint32_t)(K - kk0), (uint32_t)(M - m0), 32u, 128u, (uint64_t)K);
        // B tile (transposed): 128 N-rows x 32 K
        tdm_load_2d(ldsBo[buf], Bt + (size_t)n0 * K + kk0,
                    (uint32_t)(K - kk0), (uint32_t)(N - n0), 32u, 128u, (uint64_t)K);
    };
    if (leader) issue(0, 0);
#endif

    for (int i = 0; i < nk; i++) {
        const int cur = i & 1;
#if USE_TDM
        if (leader) {
            if (i + 1 < nk) {
                issue(i + 1, (i + 1) & 1);                  // prefetch next tile pair
                __builtin_amdgcn_s_wait_tensorcnt(2);       // current pair landed
            } else {
                __builtin_amdgcn_s_wait_tensorcnt(0);       // last pair landed
            }
        }
        __syncthreads();                                    // current tiles visible
#else
        __syncthreads();
        {
            int k0 = i * 32;
            for (int j = tid; j < 128 * 32; j += 256) {
                int rr = j >> 5, kk = j & 31;
                int gm = m0 + rr, gk = k0 + kk;
                sA[cur][j] = (gm < M && gk < K) ? A[(size_t)gm * K + gk] : (uint16_t)0;
            }
            for (int j = tid; j < 128 * 32; j += 256) {
                int rr = j >> 5, kk = j & 31;
                int gn = n0 + rr, gk = k0 + kk;
                sB[cur][j] = (gn < N && gk < K) ? Bt[(size_t)gn * K + gk] : (uint16_t)0;
            }
        }
        __syncthreads();
#endif
        union Frag { v4u q[2]; v16bf v; };
        Frag af[2], bf_[4];
        // A 16x32 operand: lane holds row M = r; lane-half g selects K groups.
        // Elements per lane are dword-contiguous: [g*8 .. g*8+7] and [16+g*8 .. +7].
#pragma unroll
        for (int t = 0; t < 2; t++) {
            const uint16_t* ap = &sA[cur][(size_t)(wm * 32 + t * 16 + r) * 32];
            af[t].q[0] = *(const v4u*)(ap + g * 8);
            af[t].q[1] = *(const v4u*)(ap + 16 + g * 8);
        }
        // B 32x16 operand: lane holds col N = r; same K grouping as A.
        // With Bt N-major in LDS: [g*8 .. g*8+7] and [16+g*8 .. +7].
#pragma unroll
        for (int u = 0; u < 4; u++) {
            const uint16_t* bp = &sB[cur][(size_t)(wn * 64 + u * 16 + r) * 32];
            bf_[u].q[0] = *(const v4u*)(bp + g * 8);
            bf_[u].q[1] = *(const v4u*)(bp + 16 + g * 8);
        }
#pragma unroll
        for (int t = 0; t < 2; t++)
#pragma unroll
            for (int u = 0; u < 4; u++)
                acc[t][u] = __builtin_amdgcn_wmma_f32_16x16x32_bf16(
                    false, af[t].v, false, bf_[u].v, (short)0, acc[t][u], false, false);
        __syncthreads();                                    // done with cur before refill
    }

    // Epilogue (C/D layout: VGPR e -> row e+8*g, col = lane&15)
#pragma unroll
    for (int t = 0; t < 2; t++) {
#pragma unroll
        for (int u = 0; u < 4; u++) {
            int col = n0 + wn * 64 + u * 16 + r;
            if (col >= N) continue;
            float cb = bias  ? bias[col]  : 0.0f;
            float sc = scale ? scale[col] : 1.0f;
            float bsc = 1.0f, bsh = 0.0f;
            if (bng) {
                float inv = rsqrtf(bnv[col] + 1e-3f);
                bsc = bng[col] * inv;
                bsh = bnb[col] - bnm[col] * bsc;
            }
#pragma unroll
            for (int e = 0; e < 8; e++) {
                int row = m0 + wm * 32 + t * 16 + g * 8 + e;
                if (row >= M) continue;
                float x = acc[t][u][e] + cb;
                if (bng)   x = x * bsc + bsh;
                if (resid) x += resid[(size_t)row * N + col];
                if (do_gelu) x = gelu_f(x);
                x *= sc;
                if (Cf)  Cf[(size_t)row * N + col]  = x;
                if (Cbf) Cbf[(size_t)row * N + col] = f2bf(x);
            }
        }
    }
}

// ---------------------------------------------------------------------------
// Support kernels
// ---------------------------------------------------------------------------
__global__ __launch_bounds__(256) void k_cvt_bf16(const float* __restrict__ in,
                                                  uint16_t* __restrict__ out, size_t n)
{
    size_t i = (size_t)blockIdx.x * 256 + threadIdx.x;
    if (i < n) out[i] = f2bf(in[i]);
}

// fp32 weight (K x N, row-major) -> bf16 transposed (N x K)
__global__ __launch_bounds__(256) void k_cvtT_bf16(const float* __restrict__ in,
                                                   uint16_t* __restrict__ out, int K, int N)
{
    size_t i = (size_t)blockIdx.x * 256 + threadIdx.x;
    if (i >= (size_t)K * N) return;
    int k = (int)(i / N);
    int n = (int)(i % N);
    out[(size_t)n * K + k] = f2bf(in[i]);
}

// stem conv1: (2,512,512,3) -> (2,256,256,96), 3x3 s2 SAME(pad_low=0), BN, GELU
__global__ __launch_bounds__(256) void k_conv_stem1(
    const float* __restrict__ in, const float* __restrict__ w,
    const float* __restrict__ gg, const float* __restrict__ bb,
    const float* __restrict__ mm, const float* __restrict__ vv,
    float* __restrict__ out)
{
    size_t i = (size_t)blockIdx.x * 256 + threadIdx.x;
    const size_t total = 2ull * 256 * 256 * 96;
    if (i >= total) return;
    int c  = (int)(i % 96);
    int ox = (int)((i / 96) % 256);
    int oy = (int)((i / (96ull * 256)) % 256);
    int b  = (int)(i / (96ull * 256 * 256));
    float s = 0.0f;
    for (int ky = 0; ky < 3; ky++) {
        int iy = oy * 2 + ky; if (iy >= 512) continue;
        for (int kx = 0; kx < 3; kx++) {
            int ix = ox * 2 + kx; if (ix >= 512) continue;
            const float* ip = in + (((size_t)b * 512 + iy) * 512 + ix) * 3;
            const float* wp = w + (size_t)((ky * 3 + kx) * 3) * 96 + c;
            s += ip[0] * wp[0] + ip[1] * wp[96] + ip[2] * wp[192];
        }
    }
    float inv = rsqrtf(vv[c] + 1e-3f);
    s = (s - mm[c]) * inv * gg[c] + bb[c];
    out[i] = gelu_f(s);
}

// im2col (3x3) into bf16 rows [m0, m0+mc) of M = Bn*Ho*Wo, K = 9*C
__global__ __launch_bounds__(256) void k_im2col_bf16(
    const float* __restrict__ src, uint16_t* __restrict__ dst,
    int Bn, int H, int W, int C, int stride, int pad, int Ho, int Wo,
    int m0, int mc)
{
    (void)Bn;
    const int K = 9 * C;
    size_t i = (size_t)blockIdx.x * 256 + threadIdx.x;
    if (i >= (size_t)mc * K) return;
    int kidx = (int)(i % K);
    int m    = (int)(i / K);
    int gm   = m0 + m;
    int ox = gm % Wo;
    int oy = (gm / Wo) % Ho;
    int b  = gm / (Wo * Ho);
    int ci = kidx % C;
    int kx = (kidx / C) % 3;
    int ky = kidx / (3 * C);
    int iy = oy * stride + ky - pad;
    int ix = ox * stride + kx - pad;
    float v = 0.0f;
    if (iy >= 0 && iy < H && ix >= 0 && ix < W)
        v = src[(((size_t)b * H + iy) * W + ix) * C + ci];
    dst[i] = f2bf(v);
}

// Fourier positional encoding: y[b,hp,wp,c] += (pe @ W + b), pe recomputed on the fly
__global__ __launch_bounds__(256) void k_posenc_add(
    const float* __restrict__ Wp, const float* __restrict__ bp, float* __restrict__ y)
{
    int i = (int)(blockIdx.x * 256 + threadIdx.x);
    const int total = 64 * 64 * 384;
    if (i >= total) return;
    int c  = i % 384;
    int wp = (i / 384) % 64;
    int hp = i / (384 * 64);
    float my = ((float)hp / 64.0f + 1e-6f) * 6.283185307f;
    float mx = ((float)wp / 64.0f + 1e-6f) * 6.283185307f;
    float s = bp[c];
    for (int j = 0; j < 64; j++) {
        int a = j >> 5, t = j & 31;
        float mesh = a ? mx : my;
        int u = (t < 16) ? t : (t - 16);
        float arg = mesh * __powf(10000.0f, -(float)u / 16.0f);
        float pe = (t < 16) ? __sinf(arg) : __cosf(arg);
        s += pe * Wp[(size_t)j * 384 + c];
    }
    y[i] += s;
    y[(size_t)total + i] += s;
}

// LayerNorm over 384; outputs optional fp32 and/or bf16
__global__ __launch_bounds__(128) void k_layernorm(
    const float* __restrict__ in, const float* __restrict__ gg, const float* __restrict__ bb,
    float* __restrict__ of32, uint16_t* __restrict__ obf, int C)
{
    const int tok = (int)blockIdx.x;
    const float* x = in + (size_t)tok * C;
    __shared__ float red[128];
    float s = 0.0f;
    for (int c = (int)threadIdx.x; c < C; c += 128) s += x[c];
    red[threadIdx.x] = s; __syncthreads();
    for (int off = 64; off > 0; off >>= 1) {
        if ((int)threadIdx.x < off) red[threadIdx.x] += red[threadIdx.x + off];
        __syncthreads();
    }
    float mu = red[0] / (float)C;
    __syncthreads();
    float s2 = 0.0f;
    for (int c = (int)threadIdx.x; c < C; c += 128) { float d = x[c] - mu; s2 += d * d; }
    red[threadIdx.x] = s2; __syncthreads();
    for (int off = 64; off > 0; off >>= 1) {
        if ((int)threadIdx.x < off) red[threadIdx.x] += red[threadIdx.x + off];
        __syncthreads();
    }
    float inv = rsqrtf(red[0] / (float)C + 1e-3f);
    for (int c = (int)threadIdx.x; c < C; c += 128) {
        float v = (x[c] - mu) * inv * gg[c] + bb[c];
        if (of32) of32[(size_t)tok * C + c] = v;
        if (obf)  obf[(size_t)tok * C + c]  = f2bf(v);
    }
}

// Split qkv (B,N,h*48*3) -> qhat/khat (L2-normalized over tokens) and v, each (B,N,384)
__global__ __launch_bounds__(256) void k_l2norm_split(
    const float* __restrict__ qkv, float* __restrict__ qh,
    float* __restrict__ kh, float* __restrict__ vb)
{
    int idx  = (int)blockIdx.x;            // (b,h,d,comp)
    int comp = idx % 3;
    int d    = (idx / 3) % 48;
    int h    = (idx / 144) % 8;
    int b    = idx / 1152;
    const int N = 4096;
    size_t srcBase = (size_t)b * N * 1152 + (size_t)h * 144 + (size_t)d * 3 + comp;
    size_t dstBase = (size_t)b * N * 384 + (size_t)h * 48 + d;
    float* dst = (comp == 0) ? qh : (comp == 1) ? kh : vb;
    __shared__ float red[256];
    float rn = 1.0f;
    if (comp < 2) {
        float s = 0.0f;
        for (int n = (int)threadIdx.x; n < N; n += 256) {
            float x = qkv[srcBase + (size_t)n * 1152];
            s += x * x;
        }
        red[threadIdx.x] = s; __syncthreads();
        for (int off = 128; off > 0; off >>= 1) {
            if ((int)threadIdx.x < off) red[threadIdx.x] += red[threadIdx.x + off];
            __syncthreads();
        }
        rn = 1.0f / sqrtf(red[0]);
    }
    for (int n = (int)threadIdx.x; n < N; n += 256)
        dst[dstBase + (size_t)n * 384] = qkv[srcBase + (size_t)n * 1152] * rn;
}

// XCA channel-covariance attention per (b,h): attn = softmax(q^T k * temp); out = attn*v
__global__ __launch_bounds__(256) void k_xca(
    const float* __restrict__ qh, const float* __restrict__ kh,
    const float* __restrict__ vb, const float* __restrict__ temp,
    float* __restrict__ out)
{
    int b = (int)blockIdx.x >> 3, h = (int)blockIdx.x & 7;
    const int N = 4096;
    __shared__ float qt[64 * 48];
    __shared__ float kt[64 * 48];
    __shared__ float attn[48 * 49];
    float accv[9];
#pragma unroll
    for (int i = 0; i < 9; i++) accv[i] = 0.0f;
    size_t base = (size_t)b * N * 384 + (size_t)h * 48;

    for (int n0 = 0; n0 < N; n0 += 64) {
        __syncthreads();
        for (int i = (int)threadIdx.x; i < 64 * 48; i += 256) {
            int t = i / 48, c = i % 48;
            qt[i] = qh[base + (size_t)(n0 + t) * 384 + c];
            kt[i] = kh[base + (size_t)(n0 + t) * 384 + c];
        }
        __syncthreads();
#pragma unroll
        for (int i = 0; i < 9; i++) {
            int e = (int)threadIdx.x + i * 256;
            if (e < 2304) {
                int c = e / 48, d = e % 48;
                float s = accv[i];
                for (int t = 0; t < 64; t++) s += qt[t * 48 + c] * kt[t * 48 + d];
                accv[i] = s;
            }
        }
    }
    __syncthreads();
    float tmp = temp[h];
#pragma unroll
    for (int i = 0; i < 9; i++) {
        int e = (int)threadIdx.x + i * 256;
        if (e < 2304) attn[(e / 48) * 49 + (e % 48)] = accv[i] * tmp;
    }
    __syncthreads();
    if ((int)threadIdx.x < 48) {
        int c = (int)threadIdx.x;
        float mx = -3.4e38f;
        for (int d = 0; d < 48; d++) mx = fmaxf(mx, attn[c * 49 + d]);
        float ssum = 0.0f;
        for (int d = 0; d < 48; d++) { float e = __expf(attn[c * 49 + d] - mx); attn[c * 49 + d] = e; ssum += e; }
        float inv = 1.0f / ssum;
        for (int d = 0; d < 48; d++) attn[c * 49 + d] *= inv;
    }
    __syncthreads();
    for (int n = (int)threadIdx.x; n < N; n += 256) {
        float vl[48];
#pragma unroll
        for (int d = 0; d < 48; d++) vl[d] = vb[base + (size_t)n * 384 + d];
#pragma unroll
        for (int c = 0; c < 48; c++) {
            float s = 0.0f;
#pragma unroll
            for (int d = 0; d < 48; d++) s += attn[c * 49 + d] * vl[d];
            out[base + (size_t)n * 384 + c] = s;
        }
    }
}

// Depthwise 3x3 on (2,64,64,384). mode 1: bn(gelu(conv+b)); mode 2: (conv+b)*scale
__global__ __launch_bounds__(256) void k_dwconv(
    const float* __restrict__ in, const float* __restrict__ w, const float* __restrict__ wb,
    const float* __restrict__ gg, const float* __restrict__ bb,
    const float* __restrict__ mm, const float* __restrict__ vv,
    const float* __restrict__ sc, float* __restrict__ out, int mode)
{
    const int H = 64, W = 64, C = 384;
    size_t i = (size_t)blockIdx.x * 256 + threadIdx.x;
    const size_t total = 2ull * H * W * C;
    if (i >= total) return;
    int c = (int)(i % C);
    int x = (int)((i / C) % W);
    int y = (int)((i / ((size_t)C * W)) % H);
    int b = (int)(i / ((size_t)C * W * H));
    float s = 0.0f;
    for (int dy = -1; dy <= 1; dy++) {
        int iy = y + dy; if (iy < 0 || iy >= H) continue;
        for (int dx = -1; dx <= 1; dx++) {
            int ix = x + dx; if (ix < 0 || ix >= W) continue;
            s += in[(((size_t)b * H + iy) * W + ix) * C + c] *
                 w[(size_t)((dy + 1) * 3 + (dx + 1)) * C + c];
        }
    }
    s += wb[c];
    if (mode == 1) {
        s = gelu_f(s);
        float inv = rsqrtf(vv[c] + 1e-3f);
        s = (s - mm[c]) * inv * gg[c] + bb[c];
    } else {
        s *= sc[c];
    }
    out[i] = s;
}

// nearest-2x upsample + fp32->bf16 gather: out rows M = Bn*2Hs*2Ws of C channels
__global__ __launch_bounds__(256) void k_up_cvt_bf16(
    const float* __restrict__ src, uint16_t* __restrict__ dst,
    int Bn, int Hs, int Ws, int C)
{
    size_t i = (size_t)blockIdx.x * 256 + threadIdx.x;
    size_t total = (size_t)Bn * (2 * Hs) * (2 * Ws) * C;
    if (i >= total) return;
    int c = (int)(i % C);
    size_t m = i / C;
    int x = (int)(m % (2 * Ws));
    int y = (int)((m / (2 * Ws)) % (2 * Hs));
    int b = (int)(m / ((size_t)4 * Hs * Ws));
    dst[i] = f2bf(src[(((size_t)b * Hs + (y >> 1)) * Ws + (x >> 1)) * C + c]);
}

// ---------------------------------------------------------------------------
// Host orchestration
// ---------------------------------------------------------------------------
static void gemm(hipStream_t s, const void* A, const void* Bt, float* Cf, uint16_t* Cbf,
                 int M, int N, int K, const float* bias,
                 const float* bng, const float* bnb, const float* bnm, const float* bnv,
                 const float* resid, const float* scale, int do_gelu)
{
    dim3 grid((unsigned)((N + 127) / 128), (unsigned)((M + 127) / 128));
    k_gemm_bf16_wmma<<<grid, dim3(256), 0, s>>>(
        (const uint16_t*)A, (const uint16_t*)Bt, Cf, Cbf, M, N, K,
        bias, bng, bnb, bnm, bnv, resid, scale, do_gelu);
}

extern "C" void kernel_launch(void* const* d_in, const int* in_sizes, int n_in,
                              void* d_out, int out_size, void* d_ws, size_t ws_size,
                              hipStream_t stream)
{
    (void)in_sizes; (void)n_in; (void)out_size; (void)ws_size;
    auto IN = [&](int i) { return (const float*)d_in[i]; };

    constexpr size_t NF0 = 2ull * 256 * 256 * 96;   // 12,582,912
    constexpr size_t NF1 = 2ull * 128 * 128 * 192;  //  6,291,456
    constexpr size_t NY  = 2ull * 64 * 64 * 384;    //  3,145,728
    constexpr size_t NQKV = 2ull * 4096 * 1152;
    constexpr size_t NH   = 2ull * 4096 * 1536;

    char* wsb = (char*)d_ws;
    size_t off = 0;
    auto take = [&](size_t bytes) { size_t r = off; off = (off + bytes + 255) & ~(size_t)255; return r; };
    // Persistent
    size_t oF0 = take(NF0 * 4), oF1 = take(NF1 * 4), oY = take(NY * 4), oWBF = take(24ull << 20);
    size_t arena = off;
    // Transformer-phase overlay
    size_t a = arena;
    auto takeT = [&](size_t bytes) { size_t r = a; a = (a + bytes + 255) & ~(size_t)255; return r; };
    size_t oY2 = takeT(NY * 4), oZ32 = takeT(NY * 4), oTMP = takeT(NY * 4), oQKV = takeT(NQKV * 4);
    size_t oQH = takeT(NY * 4), oKH = takeT(NY * 4), oVB = takeT(NY * 4), oXCA = takeT(NY * 4);
    size_t oTBF = takeT(NY * 2), oHBF = takeT(NH * 2);
    // Stem/decoder-phase overlay (reuses the same arena)
    size_t d = arena;
    auto takeD = [&](size_t bytes) { size_t r = d; d = (d + bytes + 255) & ~(size_t)255; return r; };
    size_t oUPA = takeD(131072ull * 192 * 2);
    size_t oIMC = takeD(32768ull * 1728 * 2);
    size_t oZA  = takeD(131072ull * 96 * 4);
    size_t oZB  = takeD(131072ull * 96 * 4);

    float*    F0  = (float*)(wsb + oF0);
    float*    F1  = (float*)(wsb + oF1);
    float*    Y   = (float*)(wsb + oY);
    float*    Y2  = (float*)(wsb + oY2);
    float*    Z32 = (float*)(wsb + oZ32);
    float*    TMP = (float*)(wsb + oTMP);
    float*    QKV = (float*)(wsb + oQKV);
    float*    QH  = (float*)(wsb + oQH);
    float*    KH  = (float*)(wsb + oKH);
    float*    VB  = (float*)(wsb + oVB);
    float*    XCA = (float*)(wsb + oXCA);
    uint16_t* TBF = (uint16_t*)(wsb + oTBF);
    uint16_t* HBF = (uint16_t*)(wsb + oHBF);
    uint16_t* UPA = (uint16_t*)(wsb + oUPA);
    uint16_t* IMC = (uint16_t*)(wsb + oIMC);
    float*    ZA  = (float*)(wsb + oZA);
    float*    ZB  = (float*)(wsb + oZB);

    uint16_t* wpool = (uint16_t*)(wsb + oWBF);
    // convert fp32 weight (KxN) to bf16 transposed (NxK) in the pool
    auto cvtW = [&](const float* w, int K, int N) {
        uint16_t* p = wpool;
        size_t n = (size_t)K * N;
        k_cvtT_bf16<<<(unsigned)((n + 255) / 256), 256, 0, stream>>>(w, p, K, N);
        wpool += (n + 127) & ~(size_t)127;
        return p;
    };

    // ---- stem ----
    k_conv_stem1<<<(unsigned)((NF0 + 255) / 256), 256, 0, stream>>>(
        IN(0), IN(1), IN(2), IN(3), IN(4), IN(5), F0);

    uint16_t* Wst2 = cvtW(IN(6), 864, 192);
    k_im2col_bf16<<<(unsigned)((32768ull * 864 + 255) / 256), 256, 0, stream>>>(
        F0, IMC, 2, 256, 256, 96, 2, 0, 128, 128, 0, 32768);
    gemm(stream, IMC, Wst2, F1, nullptr, 32768, 192, 864,
         nullptr, IN(7), IN(8), IN(9), IN(10), nullptr, nullptr, 1);

    uint16_t* Wst3 = cvtW(IN(11), 1728, 384);
    k_im2col_bf16<<<(unsigned)((8192ull * 1728 + 255) / 256), 256, 0, stream>>>(
        F1, IMC, 2, 128, 128, 192, 2, 0, 64, 64, 0, 8192);
    gemm(stream, IMC, Wst3, Y, nullptr, 8192, 384, 1728,
         nullptr, IN(12), IN(13), IN(14), IN(15), nullptr, nullptr, 0);

    k_posenc_add<<<(unsigned)((64 * 64 * 384 + 255) / 256), 256, 0, stream>>>(IN(16), IN(17), Y);

    // ---- XCA transformer blocks ----
    for (int blk = 0; blk < 4; blk++) {
        int p = 18 + 26 * blk;
        uint16_t* Wq  = cvtW(IN(p + 7), 384, 1152);
        uint16_t* Wpj = cvtW(IN(p + 9), 384, 384);
        uint16_t* W1b = cvtW(IN(p + 19), 384, 1536);
        uint16_t* W2b = cvtW(IN(p + 21), 1536, 384);

        k_layernorm<<<8192, 128, 0, stream>>>(Y, IN(p + 0), IN(p + 1), nullptr, TBF, 384);
        gemm(stream, TBF, Wq, QKV, nullptr, 8192, 1152, 384,
             IN(p + 8), nullptr, nullptr, nullptr, nullptr, nullptr, nullptr, 0);
        k_l2norm_split<<<2304, 256, 0, stream>>>(QKV, QH, KH, VB);
        k_xca<<<16, 256, 0, stream>>>(QH, KH, VB, IN(p + 6), XCA);
        k_cvt_bf16<<<(unsigned)((NY + 255) / 256), 256, 0, stream>>>(XCA, TBF, NY);
        gemm(stream, TBF, Wpj, Y2, nullptr, 8192, 384, 384,
             IN(p + 10), nullptr, nullptr, nullptr, nullptr, nullptr, IN(p + 23), 0);

        k_layernorm<<<8192, 128, 0, stream>>>(Y2, IN(p + 2), IN(p + 3), Z32, nullptr, 384);
        k_dwconv<<<(unsigned)((NY + 255) / 256), 256, 0, stream>>>(
            Z32, IN(p + 11), IN(p + 12), IN(p + 13), IN(p + 14), IN(p + 15), IN(p + 16),
            nullptr, TMP, 1);
        k_dwconv<<<(unsigned)((NY + 255) / 256), 256, 0, stream>>>(
            TMP, IN(p + 17), IN(p + 18), nullptr, nullptr, nullptr, nullptr,
            IN(p + 24), Y2, 2);

        k_layernorm<<<8192, 128, 0, stream>>>(Y2, IN(p + 4), IN(p + 5), nullptr, TBF, 384);
        gemm(stream, TBF, W1b, nullptr, HBF, 8192, 1536, 384,
             IN(p + 20), nullptr, nullptr, nullptr, nullptr, nullptr, nullptr, 1);
        gemm(stream, HBF, W2b, Y, nullptr, 8192, 384, 1536,
             IN(p + 22), nullptr, nullptr, nullptr, nullptr, nullptr, IN(p + 25), 0);
    }

    // ---- outputs / decoder ----
    float* OUT0 = (float*)d_out;
    float* OUT1 = OUT0 + NF0;
    float* OUT2 = OUT1 + NF1;
    (void)hipMemcpyAsync(OUT2, Y, NY * sizeof(float), hipMemcpyDeviceToDevice, stream);

    // k = 2: up(feats2') -> mix1(384->192) + F1 -> conv192 x2 (gelu)
    uint16_t* Wm1  = cvtW(IN(124), 384, 192);
    uint16_t* Wc10 = cvtW(IN(130), 1728, 192);
    uint16_t* Wc11 = cvtW(IN(132), 1728, 192);
    k_up_cvt_bf16<<<(unsigned)((32768ull * 384 + 255) / 256), 256, 0, stream>>>(Y, UPA, 2, 64, 64, 384);
    gemm(stream, UPA, Wm1, ZA, nullptr, 32768, 192, 384,
         IN(125), nullptr, nullptr, nullptr, nullptr, F1, nullptr, 0);
    k_im2col_bf16<<<(unsigned)((32768ull * 1728 + 255) / 256), 256, 0, stream>>>(
        ZA, IMC, 2, 128, 128, 192, 1, 1, 128, 128, 0, 32768);
    gemm(stream, IMC, Wc10, ZB, nullptr, 32768, 192, 1728,
         IN(131), nullptr, nullptr, nullptr, nullptr, nullptr, nullptr, 1);
    k_im2col_bf16<<<(unsigned)((32768ull * 1728 + 255) / 256), 256, 0, stream>>>(
        ZB, IMC, 2, 128, 128, 192, 1, 1, 128, 128, 0, 32768);
    gemm(stream, IMC, Wc11, OUT1, nullptr, 32768, 192, 1728,
         IN(133), nullptr, nullptr, nullptr, nullptr, nullptr, nullptr, 1);

    // k = 1: up(F1) -> mix0(192->96) + F0 -> conv96 x2 (gelu), M chunked by 32768
    uint16_t* Wm0  = cvtW(IN(122), 192, 96);
    uint16_t* Wc00 = cvtW(IN(126), 864, 96);
    uint16_t* Wc01 = cvtW(IN(128), 864, 96);
    k_up_cvt_bf16<<<(unsigned)((131072ull * 192 + 255) / 256), 256, 0, stream>>>(F1, UPA, 2, 128, 128, 192);
    gemm(stream, UPA, Wm0, ZA, nullptr, 131072, 96, 192,
         IN(123), nullptr, nullptr, nullptr, nullptr, F0, nullptr, 0);
    for (int ch = 0; ch < 4; ch++) {
        int m0 = ch * 32768;
        k_im2col_bf16<<<(unsigned)((32768ull * 864 + 255) / 256), 256, 0, stream>>>(
            ZA, IMC, 2, 256, 256, 96, 1, 1, 256, 256, m0, 32768);
        gemm(stream, IMC, Wc00, ZB + (size_t)m0 * 96, nullptr, 32768, 96, 864,
             IN(127), nullptr, nullptr, nullptr, nullptr, nullptr, nullptr, 1);
    }
    for (int ch = 0; ch < 4; ch++) {
        int m0 = ch * 32768;
        k_im2col_bf16<<<(unsigned)((32768ull * 864 + 255) / 256), 256, 0, stream>>>(
            ZB, IMC, 2, 256, 256, 96, 1, 1, 256, 256, m0, 32768);
        gemm(stream, IMC, Wc01, OUT0 + (size_t)m0 * 96, nullptr, 32768, 96, 864,
             IN(129), nullptr, nullptr, nullptr, nullptr, nullptr, nullptr, 1);
    }
}
